// Flowformer_89455578841815
// MI455X (gfx1250) — matrix operand order repaired and checked by reading, not compiled, not run on hardware
//
#include <hip/hip_runtime.h>
#include <cstdint>
#include <cstddef>

// ---------------- problem constants ----------------
#define B_    4
#define L_    2048
#define CIN_  64
#define COUT_ 64
#define D_    1024
#define H_    16
#define DH_   64
#define F_    4096
#define E_    4
#define M_    (B_ * L_)     // 8192 tokens

typedef unsigned short u16;
typedef __attribute__((ext_vector_type(16))) __bf16 v16bf;
typedef __attribute__((ext_vector_type(8)))  float  v8f;

#ifndef __has_builtin
#define __has_builtin(x) 0
#endif
#if __has_builtin(__builtin_amdgcn_global_load_async_to_lds_b128)
#define FF_ASYNC_BUILTIN 1
#else
#define FF_ASYNC_BUILTIN 0
#endif
#if __has_builtin(__builtin_amdgcn_s_wait_asynccnt)
#define FF_ASYNCCNT_BUILTIN 1
#else
#define FF_ASYNCCNT_BUILTIN 0
#endif

// exact parameter types of the async-to-LDS builtin (from clang diagnostic):
//   int4 __device__*  (global, AS1) source ; int4 __shared__* (AS3) LDS dest
typedef int ff_v4i __attribute__((vector_size(16)));
typedef __attribute__((address_space(1))) ff_v4i* ff_gptr;
typedef __attribute__((address_space(3))) ff_v4i* ff_lptr;

__device__ __forceinline__ u16 f2bf(float x) {
    union { float f; unsigned u; } c; c.f = x;
    unsigned u = c.u;
    u += 0x7FFFu + ((u >> 16) & 1u);   // round-to-nearest-even
    return (u16)(u >> 16);
}
__device__ __forceinline__ float sigf(float x) { return 1.f / (1.f + __expf(-x)); }

// 16B async copy global -> LDS (no VGPR round-trip, tracked by ASYNCcnt)
__device__ __forceinline__ void async_ld_b128(const u16* g, u16* l) {
#if FF_ASYNC_BUILTIN
    __builtin_amdgcn_global_load_async_to_lds_b128((ff_gptr)g, (ff_lptr)l, 0, 0);
#else
    asm volatile("global_load_async_to_lds_b128 %0, %1, off"
                 :: "v"((unsigned)(uintptr_t)l), "v"(g)
                 : "memory");
#endif
}
__device__ __forceinline__ void wait_async0() {
#if FF_ASYNCCNT_BUILTIN
    __builtin_amdgcn_s_wait_asynccnt(0);
#else
    asm volatile("s_wait_asynccnt 0x0" ::: "memory");
#endif
}

// =====================================================================
// WMMA GEMM:  C[M,N] = act(A[M,K](bf16) * W[N,K]^T(bf16) + bias[N])
// Tile 128x128x32, 256 threads = 8 waves (2x4), wave = 64x32 output.
// Ping-pong LDS buffers filled by GLOBAL_LOAD_ASYNC_TO_LDS_B128,
// consumed by v_wmma_f32_16x16x32_bf16.
// Requires: M % 128 == 0, K % 32 == 0. N edge handled (zero-fill + guard).
// =====================================================================
#define GBM 128
#define GBN 128
#define GBK 32
#define LDK 40   // padded row stride in u16 (80B, keeps 16B alignment)

__global__ __launch_bounds__(256, 2)
void ff_gemm_bf16(const u16* __restrict__ A, const u16* __restrict__ W,
                  const float* __restrict__ bias,
                  float* __restrict__ outF, u16* __restrict__ outH,
                  int M, int N, int K, int relu)
{
    __shared__ u16 As0[GBM * LDK];
    __shared__ u16 As1[GBM * LDK];
    __shared__ u16 Ws0[GBN * LDK];
    __shared__ u16 Ws1[GBN * LDK];

    const int tid  = threadIdx.x;
    const int bm   = blockIdx.y * GBM;
    const int bn   = blockIdx.x * GBN;
    const int wave = tid >> 5;
    const int lane = tid & 31;
    const int wm   = (wave >> 2) * 64;   // 0 / 64
    const int wn   = (wave & 3) * 32;    // 0 / 32 / 64 / 96
    const int half = lane >> 4;          // 0: lanes 0-15, 1: lanes 16-31
    const int l16  = lane & 15;

    v8f acc[4][2];
    #pragma unroll
    for (int i = 0; i < 4; ++i)
        #pragma unroll
        for (int j = 0; j < 2; ++j)
            #pragma unroll
            for (int g = 0; g < 8; ++g) acc[i][j][g] = 0.f;

    // issue async staging of one 128x32 A tile + 128x32 W tile into (Ab,Wb)
    auto stageTo = [&](u16* Ab, u16* Wb, int k0) {
        if (k0 + 2 * GBK < K)   // L2 prefetch two tiles ahead
            __builtin_prefetch(A + (size_t)(bm + (tid >> 1)) * K + k0 + 2 * GBK, 0, 1);
        #pragma unroll
        for (int it = 0; it < 2; ++it) {
            int idx = tid + it * 256;
            int r = idx >> 2, c = (idx & 3) << 3;      // 8 u16 per 16B chunk
            async_ld_b128(A + (size_t)(bm + r) * K + k0 + c, &Ab[r * LDK + c]);
        }
        #pragma unroll
        for (int it = 0; it < 2; ++it) {
            int idx = tid + it * 256;
            int r = idx >> 2, c = (idx & 3) << 3;
            if (bn + r < N)
                async_ld_b128(W + (size_t)(bn + r) * K + k0 + c, &Wb[r * LDK + c]);
            else
                *(uint4*)(&Wb[r * LDK + c]) = make_uint4(0u, 0u, 0u, 0u);
        }
    };

    // consume one staged tile pair: 8 x v_wmma_f32_16x16x32_bf16 per wave
    auto compute = [&](const u16* Ab, const u16* Wb) {
        v16bf bfrag[2];
        #pragma unroll
        for (int nf = 0; nf < 2; ++nf) {
            // B frag: lane<16 -> col l16, K 0..15 ; lane>=16 -> K 16..31
            const u16* p = &Wb[(wn + nf * 16 + l16) * LDK + half * 16];
            union { v16bf v; uint4 q[2]; } u;
            u.q[0] = *(const uint4*)(p);
            u.q[1] = *(const uint4*)(p + 8);
            bfrag[nf] = u.v;
        }
        #pragma unroll
        for (int mf = 0; mf < 4; ++mf) {
            // A frag: lane<16 -> row l16, K 0..15 ; lane>=16 -> K 16..31
            const u16* p = &Ab[(wm + mf * 16 + l16) * LDK + half * 16];
            union { v16bf v; uint4 q[2]; } u;
            u.q[0] = *(const uint4*)(p);
            u.q[1] = *(const uint4*)(p + 8);
            v16bf afrag = u.v;
            #pragma unroll
            for (int nf = 0; nf < 2; ++nf) {
                acc[mf][nf] = __builtin_amdgcn_wmma_f32_16x16x32_bf16(
                    false, afrag, false, bfrag[nf], (short)0, acc[mf][nf], false, false);
            }
        }
    };

    const int nsteps = K / GBK;
    stageTo(As0, Ws0, 0);
    wait_async0();
    __syncthreads();

    int ks = 0;
    for (;;) {
        if (ks + 1 < nsteps) stageTo(As1, Ws1, (ks + 1) * GBK);  // prefetch pong
        compute(As0, Ws0);                                        // eat ping
        wait_async0();
        __syncthreads();
        if (++ks >= nsteps) break;

        if (ks + 1 < nsteps) stageTo(As0, Ws0, (ks + 1) * GBK);  // prefetch ping
        compute(As1, Ws1);                                        // eat pong
        wait_async0();
        __syncthreads();
        if (++ks >= nsteps) break;
    }

    // Store: D layout -> VGPR g holds (M=g, lanes 0-15) / (M=g+8, lanes 16-31), N=l16
    #pragma unroll
    for (int mf = 0; mf < 4; ++mf) {
        #pragma unroll
        for (int nf = 0; nf < 2; ++nf) {
            const int col = bn + wn + nf * 16 + l16;
            if (col >= N) continue;
            const float bval = bias ? bias[col] : 0.f;
            #pragma unroll
            for (int g = 0; g < 8; ++g) {
                const int row = bm + wm + mf * 16 + g + half * 8;
                float v = acc[mf][nf][g] + bval;
                if (relu) v = fmaxf(v, 0.f);
                const size_t o = (size_t)row * N + col;
                if (outF) outF[o] = v;
                if (outH) outH[o] = f2bf(v);
            }
        }
    }
}

// =====================================================================
// Elementwise / reduction kernels (f32)
// =====================================================================
__global__ void ff_cvt(const float* __restrict__ s, u16* __restrict__ d, size_t n) {
    size_t i = (size_t)blockIdx.x * blockDim.x + threadIdx.x;
    if (i < n) d[i] = f2bf(s[i]);
}

__global__ void ff_sigmoid(float* __restrict__ p, size_t n) {
    size_t i = (size_t)blockIdx.x * blockDim.x + threadIdx.x;
    if (i < n) p[i] = sigf(p[i]);
}

// circular im2col for token conv1d k=3: xe[m, cin*3+t] = x[b, (l+t-1)%L, cin]
__global__ void ff_im2col(const float* __restrict__ x, u16* __restrict__ xe) {
    size_t i = (size_t)blockIdx.x * blockDim.x + threadIdx.x;
    if (i >= (size_t)M_ * 192) return;
    int kk = (int)(i % 192);
    size_t m = i / 192;
    int cin = kk / 3, t = kk % 3;
    int b = (int)(m / L_), l = (int)(m % L_);
    int ls = (l + t - 1 + L_) % L_;
    xe[i] = f2bf(x[((size_t)b * L_ + ls) * CIN_ + cin]);
}

// dst[b,h,d] = sum_l src[b,l,h,d] * (wgt ? wgt[b,h,l] : 1)   (64 threads/block)
__global__ void ff_colsum(const float* __restrict__ src, const float* __restrict__ wgt,
                          float* __restrict__ dst) {
    int bh = blockIdx.x, b = bh / H_, h = bh % H_, d = threadIdx.x;
    float s = 0.f;
    for (int l = 0; l < L_; ++l) {
        float v = src[(((size_t)b * L_ + l) * H_ + h) * DH_ + d];
        s += wgt ? v * wgt[(size_t)bh * L_ + l] : v;
    }
    dst[(size_t)bh * DH_ + d] = s;
}

// out[b,h,l] = f( sum_d (src[b,l,h,d]+eps)*(vec[b,h,d]+eps) )
// mode 0: 1/x   mode 1: sigmoid(x*scale)   mode 2: x
__global__ void ff_dotvec(const float* __restrict__ src, const float* __restrict__ vec,
                          float* __restrict__ out, int mode, float scale) {
    int wid  = (int)(((size_t)blockIdx.x * blockDim.x + threadIdx.x) >> 5);
    int lane = threadIdx.x & 31;
    int l = wid % L_, bh = wid / L_, b = bh / H_, h = bh % H_;
    const float eps = 1e-6f;
    float s = 0.f;
    #pragma unroll
    for (int j = 0; j < 2; ++j) {
        int d = lane + j * 32;
        s += (src[(((size_t)b * L_ + l) * H_ + h) * DH_ + d] + eps) *
             (vec[(size_t)bh * DH_ + d] + eps);
    }
    #pragma unroll
    for (int o = 16; o > 0; o >>= 1) s += __shfl_down(s, o);
    if (lane == 0) {
        float r = (mode == 0) ? 1.f / s : (mode == 1) ? sigf(s * scale) : s;
        out[(size_t)bh * L_ + l] = r;
    }
}

// row softmax over L, scaled by S (one block per (b,h))
__global__ void ff_softmax_scale(float* __restrict__ p, float S) {
    int bh = blockIdx.x, t = threadIdx.x;
    float* row = p + (size_t)bh * L_;
    __shared__ float rm[8], rsum[8];
    float mx = -1e30f;
    for (int c = t; c < L_; c += 256) mx = fmaxf(mx, row[c]);
    #pragma unroll
    for (int o = 16; o > 0; o >>= 1) mx = fmaxf(mx, __shfl_down(mx, o));
    if ((t & 31) == 0) rm[t >> 5] = mx;
    __syncthreads();
    if (t == 0) { float m = rm[0]; for (int i = 1; i < 8; ++i) m = fmaxf(m, rm[i]); rm[0] = m; }
    __syncthreads();
    mx = rm[0];
    float sum = 0.f;
    for (int c = t; c < L_; c += 256) { float e = __expf(row[c] - mx); row[c] = e; sum += e; }
    #pragma unroll
    for (int o = 16; o > 0; o >>= 1) sum += __shfl_down(sum, o);
    if ((t & 31) == 0) rsum[t >> 5] = sum;
    __syncthreads();
    if (t == 0) { float s2 = 0; for (int i = 0; i < 8; ++i) s2 += rsum[i]; rsum[0] = s2; }
    __syncthreads();
    float inv = S / rsum[0];
    for (int c = t; c < L_; c += 256) row[c] *= inv;
}

// kv[b,h,d,e] = sum_s k[b,s,h,d] * v[b,s,h,e] * ncr[b,h,s]   (block per (b,h))
__global__ void ff_kv(const float* __restrict__ k, const float* __restrict__ v,
                      const float* __restrict__ ncr, float* __restrict__ kv) {
    int bh = blockIdx.x, b = bh / H_, h = bh % H_, t = threadIdx.x;
    const int CS = 32;
    __shared__ float kc[CS][DH_];
    __shared__ float vc[CS][DH_];
    int d = t >> 2, e0 = (t & 3) << 4;
    float acc[16];
    #pragma unroll
    for (int j = 0; j < 16; ++j) acc[j] = 0.f;
    for (int s0 = 0; s0 < L_; s0 += CS) {
        __syncthreads();
        #pragma unroll
        for (int j = 0; j < 8; ++j) {
            int idx = t + j * 256;            // 0..2047
            int ss = idx >> 6, dd = idx & 63;
            int s = s0 + ss;
            float w = ncr[(size_t)bh * L_ + s];
            size_t base = (((size_t)b * L_ + s) * H_ + h) * DH_ + dd;
            kc[ss][dd] = k[base];
            vc[ss][dd] = v[base] * w;
        }
        __syncthreads();
        for (int ss = 0; ss < CS; ++ss) {
            float kd = kc[ss][d];
            #pragma unroll
            for (int j = 0; j < 16; ++j) acc[j] += kd * vc[ss][e0 + j];
        }
    }
    #pragma unroll
    for (int j = 0; j < 16; ++j)
        kv[((size_t)bh * DH_ + d) * DH_ + e0 + j] = acc[j];
}

// out_bf16[b,l,h,e] = (sum_d q[b,l,h,d]*kv[b,h,d,e]) * nr[b,h,l] * nrr[b,h,l]
__global__ void ff_attn_out(const float* __restrict__ q, const float* __restrict__ kvm,
                            const float* __restrict__ nr, const float* __restrict__ nrr,
                            u16* __restrict__ outH) {
    const int LB = 16;
    int lc = blockIdx.x % (L_ / LB);
    int bh = blockIdx.x / (L_ / LB);
    int b = bh / H_, h = bh % H_, t = threadIdx.x;
    __shared__ float kvs[DH_][DH_];
    __shared__ float qsl[LB][DH_];
    #pragma unroll
    for (int j = 0; j < 16; ++j) {
        int idx = t + j * 256;
        kvs[idx >> 6][idx & 63] = kvm[(size_t)bh * (DH_ * DH_) + idx];
    }
    int l0 = lc * LB;
    #pragma unroll
    for (int j = 0; j < 4; ++j) {
        int idx = t + j * 256;
        int lr = idx >> 6, dd = idx & 63;
        qsl[lr][dd] = q[(((size_t)b * L_ + l0 + lr) * H_ + h) * DH_ + dd];
    }
    __syncthreads();
    int lr = t >> 4, e0 = (t & 15) << 2;
    float acc[4] = {0.f, 0.f, 0.f, 0.f};
    for (int d = 0; d < DH_; ++d) {
        float qv = qsl[lr][d];
        #pragma unroll
        for (int j = 0; j < 4; ++j) acc[j] += qv * kvs[d][e0 + j];
    }
    float sc = nr[(size_t)bh * L_ + l0 + lr] * nrr[(size_t)bh * L_ + l0 + lr];
    #pragma unroll
    for (int j = 0; j < 4; ++j)
        outH[(((size_t)b * L_ + l0 + lr) * H_ + h) * DH_ + e0 + j] = f2bf(acc[j] * sc);
}

// out = LayerNorm(a (+ b)) * w + bias ; optional f32 and bf16 outputs (block per row)
__global__ void ff_add_ln(const float* __restrict__ a, const float* __restrict__ b,
                          const float* __restrict__ w, const float* __restrict__ bias,
                          float* __restrict__ outF, u16* __restrict__ outH) {
    int row = blockIdx.x, t = threadIdx.x;
    __shared__ float rs[8], rs2[8];
    float vals[4], s = 0.f, s2 = 0.f;
    #pragma unroll
    for (int j = 0; j < 4; ++j) {
        int c = t + j * 256;
        float v = a[(size_t)row * D_ + c] + (b ? b[(size_t)row * D_ + c] : 0.f);
        vals[j] = v; s += v; s2 += v * v;
    }
    #pragma unroll
    for (int o = 16; o > 0; o >>= 1) { s += __shfl_down(s, o); s2 += __shfl_down(s2, o); }
    if ((t & 31) == 0) { rs[t >> 5] = s; rs2[t >> 5] = s2; }
    __syncthreads();
    if (t == 0) {
        float S = 0, S2 = 0;
        for (int i = 0; i < 8; ++i) { S += rs[i]; S2 += rs2[i]; }
        rs[0] = S; rs2[0] = S2;
    }
    __syncthreads();
    float mean = rs[0] / (float)D_;
    float var  = rs2[0] / (float)D_ - mean * mean;
    float inv  = rsqrtf(var + 1e-5f);
    #pragma unroll
    for (int j = 0; j < 4; ++j) {
        int c = t + j * 256;
        float y = (vals[j] - mean) * inv * w[c] + bias[c];
        if (outF) outF[(size_t)row * D_ + c] = y;
        if (outH) outH[(size_t)row * D_ + c] = f2bf(y);
    }
}

// channel attention: mean over L, 1D conv k=5 over channels + sigmoid, scale
__global__ void ff_chmean(const float* __restrict__ h, float* __restrict__ m) {
    int idx = blockIdx.x * blockDim.x + threadIdx.x;   // B*D
    int b = idx / D_, c = idx % D_;
    float s = 0.f;
    for (int l = 0; l < L_; ++l) s += h[((size_t)b * L_ + l) * D_ + c];
    m[idx] = s / (float)L_;
}
__global__ void ff_chgate(const float* __restrict__ m, const float* __restrict__ w,
                          float* __restrict__ g) {
    int idx = blockIdx.x * blockDim.x + threadIdx.x;   // B*D
    int b = idx / D_, c = idx % D_;
    float s = 0.f;
    #pragma unroll
    for (int t = 0; t < 5; ++t) {
        int cc = c + t - 2;
        if (cc >= 0 && cc < D_) s += m[(size_t)b * D_ + cc] * w[t];
    }
    g[idx] = sigf(s);
}
__global__ void ff_chscale(float* __restrict__ h, const float* __restrict__ g,
                           u16* __restrict__ hb) {
    size_t idx = (size_t)blockIdx.x * blockDim.x + threadIdx.x;   // M*D
    int c = (int)(idx % D_);
    int b = (int)(idx / ((size_t)L_ * D_));
    float v = h[idx] * g[(size_t)b * D_ + c];
    h[idx] = v;
    hb[idx] = f2bf(v);
}

// =====================================================================
// Host launch
// =====================================================================
extern "C" void kernel_launch(void* const* d_in, const int* in_sizes, int n_in,
                              void* d_out, int out_size, void* d_ws, size_t ws_size,
                              hipStream_t stream) {
    (void)in_sizes; (void)n_in; (void)out_size; (void)ws_size;

    const float* x    = (const float*)d_in[0];
    const float* tokw = (const float*)d_in[1];
    const float* Wq   = (const float*)d_in[2];
    const float* bq   = (const float*)d_in[3];
    const float* Wk   = (const float*)d_in[4];
    const float* bk   = (const float*)d_in[5];
    const float* Wv   = (const float*)d_in[6];
    const float* bv   = (const float*)d_in[7];
    const float* Wo   = (const float*)d_in[8];
    const float* bo   = (const float*)d_in[9];
    const float* c1w  = (const float*)d_in[10];
    const float* c1b  = (const float*)d_in[11];
    const float* c2w  = (const float*)d_in[12];
    const float* c2b  = (const float*)d_in[13];
    const float* ln1w = (const float*)d_in[14];
    const float* ln1b = (const float*)d_in[15];
    const float* ln2w = (const float*)d_in[16];
    const float* ln2b = (const float*)d_in[17];
    const float* caw  = (const float*)d_in[18];
    const float* lnfw = (const float*)d_in[19];
    const float* lnfb = (const float*)d_in[20];
    const float* pw   = (const float*)d_in[21];
    const float* pb   = (const float*)d_in[22];

    char* wsp = (char*)d_ws;
    size_t off = 0;
    auto alloc = [&](size_t bytes) -> void* {
        void* p = wsp + off;
        off += (bytes + 255) & ~(size_t)255;
        return p;
    };

    const size_t DD = (size_t)D_ * D_, FD = (size_t)F_ * D_, MD = (size_t)M_ * D_;

    u16* wq_bf   = (u16*)alloc(E_ * DD * 2);
    u16* wk_bf   = (u16*)alloc(E_ * DD * 2);
    u16* wv_bf   = (u16*)alloc(E_ * DD * 2);
    u16* wo_bf   = (u16*)alloc(E_ * DD * 2);
    u16* w1_bf   = (u16*)alloc(E_ * FD * 2);
    u16* w2_bf   = (u16*)alloc(E_ * FD * 2);
    u16* tokw_bf = (u16*)alloc((size_t)D_ * 192 * 2);
    u16* pw_bf   = (u16*)alloc((size_t)COUT_ * D_ * 2);

    float* h   = (float*)alloc(MD * 4);
    float* x1  = (float*)alloc(MD * 4);
    float* t0  = (float*)alloc(MD * 4);
    float* qb  = (float*)alloc(MD * 4);
    float* kb  = (float*)alloc(MD * 4);
    float* vb  = (float*)alloc(MD * 4);
    u16*   hb  = (u16*)alloc(MD * 2);
    u16*   yb  = (u16*)alloc((size_t)M_ * F_ * 2);
    u16*   xe  = (u16*)alloc((size_t)M_ * 192 * 2);

    const int BH = B_ * H_;
    float* qs_  = (float*)alloc(BH * DH_ * 4);
    float* ks_  = (float*)alloc(BH * DH_ * 4);
    float* t1_  = (float*)alloc(BH * DH_ * 4);
    float* t2_  = (float*)alloc(BH * DH_ * 4);
    float* nr_  = (float*)alloc((size_t)BH * L_ * 4);
    float* nc_  = (float*)alloc((size_t)BH * L_ * 4);
    float* nrr_ = (float*)alloc((size_t)BH * L_ * 4);
    float* ncr_ = (float*)alloc((size_t)BH * L_ * 4);
    float* kv_  = (float*)alloc((size_t)BH * DH_ * DH_ * 4);
    float* chm  = (float*)alloc((size_t)B_ * D_ * 4);
    float* chg  = (float*)alloc((size_t)B_ * D_ * 4);

    auto cvt = [&](const float* s, u16* d, size_t n) {
        ff_cvt<<<dim3((unsigned)((n + 255) / 256)), dim3(256), 0, stream>>>(s, d, n);
    };
    auto gemm = [&](const u16* A, const u16* W, const float* bias,
                    float* oF, u16* oH, int Mm, int Nn, int Kk, int relu) {
        dim3 g((Nn + GBN - 1) / GBN, Mm / GBM);
        ff_gemm_bf16<<<g, dim3(256), 0, stream>>>(A, W, bias, oF, oH, Mm, Nn, Kk, relu);
    };

    // ---- weights -> bf16 ----
    cvt(Wq,  wq_bf,   E_ * DD);
    cvt(Wk,  wk_bf,   E_ * DD);
    cvt(Wv,  wv_bf,   E_ * DD);
    cvt(Wo,  wo_bf,   E_ * DD);
    cvt(c1w, w1_bf,   E_ * FD);
    cvt(c2w, w2_bf,   E_ * FD);
    cvt(tokw, tokw_bf, (size_t)D_ * 192);
    cvt(pw,  pw_bf,   (size_t)COUT_ * D_);

    // ---- token embedding (circular conv k=3 as im2col GEMM) ----
    ff_im2col<<<dim3((M_ * 192 + 255) / 256), dim3(256), 0, stream>>>(x, xe);
    gemm(xe, tokw_bf, nullptr, h, hb, M_, D_, 192, 0);

    const int dvBlocks = (BH * L_) / 8;   // 8 waves per 256-thread block

    // ---- 4 encoder layers (+channel att) then layer 3 re-applied ----
    for (int it = 0; it < 5; ++it) {
        const int i = (it < 4) ? it : (E_ - 1);
        const u16* wqi = wq_bf + (size_t)i * DD;
        const u16* wki = wk_bf + (size_t)i * DD;
        const u16* wvi = wv_bf + (size_t)i * DD;
        const u16* woi = wo_bf + (size_t)i * DD;
        const u16* w1i = w1_bf + (size_t)i * FD;
        const u16* w2i = w2_bf + (size_t)i * FD;

        gemm(hb, wqi, bq + (size_t)i * D_, qb, nullptr, M_, D_, D_, 0);
        gemm(hb, wki, bk + (size_t)i * D_, kb, nullptr, M_, D_, D_, 0);
        gemm(hb, wvi, bv + (size_t)i * D_, vb, nullptr, M_, D_, D_, 0);

        ff_sigmoid<<<dim3((unsigned)((MD + 255) / 256)), dim3(256), 0, stream>>>(qb, MD);
        ff_sigmoid<<<dim3((unsigned)((MD + 255) / 256)), dim3(256), 0, stream>>>(kb, MD);

        ff_colsum<<<dim3(BH), dim3(DH_), 0, stream>>>(qb, nullptr, qs_);
        ff_colsum<<<dim3(BH), dim3(DH_), 0, stream>>>(kb, nullptr, ks_);
        ff_dotvec<<<dim3(dvBlocks), dim3(256), 0, stream>>>(qb, ks_, nr_, 0, 1.f);
        ff_dotvec<<<dim3(dvBlocks), dim3(256), 0, stream>>>(kb, qs_, nc_, 0, 1.f);
        ff_colsum<<<dim3(BH), dim3(DH_), 0, stream>>>(kb, nc_, t1_);
        ff_colsum<<<dim3(BH), dim3(DH_), 0, stream>>>(qb, nr_, t2_);
        ff_dotvec<<<dim3(dvBlocks), dim3(256), 0, stream>>>(qb, t1_, nrr_, 1, 1.f);
        ff_dotvec<<<dim3(dvBlocks), dim3(256), 0, stream>>>(kb, t2_, ncr_, 2, 1.f);
        ff_softmax_scale<<<dim3(BH), dim3(256), 0, stream>>>(ncr_, (float)L_);
        ff_kv<<<dim3(BH), dim3(256), 0, stream>>>(kb, vb, ncr_, kv_);
        ff_attn_out<<<dim3(BH * (L_ / 16)), dim3(256), 0, stream>>>(qb, kv_, nr_, nrr_, hb);

        gemm(hb, woi, bo + (size_t)i * D_, t0, nullptr, M_, D_, D_, 0);
        ff_add_ln<<<dim3(M_), dim3(256), 0, stream>>>(h, t0, ln1w + (size_t)i * D_,
                                                      ln1b + (size_t)i * D_, x1, hb);
        gemm(hb, w1i, c1b + (size_t)i * F_, nullptr, yb, M_, F_, D_, 1);
        gemm(yb, w2i, c2b + (size_t)i * D_, t0, nullptr, M_, D_, F_, 0);
        ff_add_ln<<<dim3(M_), dim3(256), 0, stream>>>(x1, t0, ln2w + (size_t)i * D_,
                                                      ln2b + (size_t)i * D_, h, hb);

        if (it < 4) {
            ff_chmean<<<dim3((B_ * D_) / 256), dim3(256), 0, stream>>>(h, chm);
            ff_chgate<<<dim3((B_ * D_) / 256), dim3(256), 0, stream>>>(chm, caw + (size_t)i * 5, chg);
            ff_chscale<<<dim3((unsigned)(MD / 256)), dim3(256), 0, stream>>>(h, chg, hb);
        }
    }

    // ---- final layernorm + projection ----
    ff_add_ln<<<dim3(M_), dim3(256), 0, stream>>>(h, nullptr, lnfw, lnfb, nullptr, hb);
    gemm(hb, pw_bf, pb, (float*)d_out, nullptr, M_, COUT_, D_, 0);
}
